// TreeLSTMNet_52828097741162
// MI455X (gfx1250) — compile-verified, matrix operand order checked by reading, not run on hardware
//
#include <hip/hip_runtime.h>
#include <hip/hip_bf16.h>
#include <math.h>

#define N_NODES 100000
#define N_EDGES 100000
#define NB      256
#define XDIM    256
#define HD      256
#define NL      3
#define CONDD   256
#define EGOD    64
#define MHD     256
#define NHD     (N_NODES * HD)      // 25,600,000 elements
#define OUTSTR  (CONDD + EGOD)      // 320

typedef __attribute__((ext_vector_type(16))) __bf16 v16bf;
typedef __attribute__((ext_vector_type(8)))  float  v8f;
typedef __attribute__((ext_vector_type(4)))  unsigned int v4u32;
typedef __attribute__((ext_vector_type(8)))  int    v8i32;
typedef __attribute__((ext_vector_type(4)))  int    v4i32;

union FragBF { v16bf v; unsigned int u[8]; };

__device__ __forceinline__ unsigned short f2bf(float f) {
    unsigned int u = __float_as_uint(f);
    unsigned int r = u + 0x7fffu + ((u >> 16) & 1u);   // round-to-nearest-even
    return (unsigned short)(r >> 16);
}

// ---------------- Tensor Data Mover: DMA one 16x256 bf16 tile into LDS ----------------
// LDS layout gets +8 bf16 pad per 256-element row: pad_interval code 6 (128 DWORDs),
// pad_amount code 3 (4 DWORDs). D# per CDNA5 ISA.
__device__ __forceinline__ void tdm_load_tile_bf16(const unsigned short* gptr,
                                                   unsigned int lds_off) {
    unsigned long long ga = (unsigned long long)gptr;
    v4u32 g0;
    g0.x = 1u;                                               // count=1 (valid), no gather
    g0.y = lds_off;                                          // lds_addr
    g0.z = (unsigned int)(ga & 0xFFFFFFFFu);                 // global_addr[31:0]
    g0.w = (unsigned int)((ga >> 32) & 0x1FFFFFFu)           // global_addr[56:32]
         | (2u << 30);                                       // type = 2 ("image")
    v8i32 g1;
    g1[0] = (1 << 16)            // data_size = 2 bytes
          | (1 << 20)            // pad_enable
          | (6 << 22)            // pad_interval: 128 DWORDs (one 512B row)
          | (3 << 25);           // pad_amount: 4 DWORDs (8 bf16)
    g1[1] = (int)(256u << 16);   // tensor_dim0[15:0] = 256
    g1[2] = (int)(16u  << 16);   // tensor_dim0[31:16]=0 | tensor_dim1[15:0] = 16
    g1[3] = (int)(256u << 16);   // tensor_dim1[31:16]=0 | tile_dim0 = 256
    g1[4] = 16;                  // tile_dim1 = 16, tile_dim2 = 0
    g1[5] = 256;                 // tensor_dim0_stride[31:0] = 256 elements
    g1[6] = 0;
    g1[7] = 0;
    v4i32 z4 = {0, 0, 0, 0};
#if defined(__clang_major__) && (__clang_major__ >= 23)
    v8i32 z8 = {0, 0, 0, 0, 0, 0, 0, 0};
    __builtin_amdgcn_tensor_load_to_lds(g0, g1, z4, z4, z8, 0);
#else
    __builtin_amdgcn_tensor_load_to_lds(g0, g1, z4, z4, 0);
#endif
}

// ---------------- f32 -> bf16 conversion (weights, x mirror) ----------------
__global__ void cvt_bf16_kernel(const float* __restrict__ w, unsigned short* __restrict__ o, int n) {
    int i = blockIdx.x * blockDim.x + threadIdx.x;
    if (i < n) o[i] = f2bf(w[i]);
}

// ---------------- zero fill ----------------
__global__ void zero_f32_kernel(float* __restrict__ p, int n) {
    int stride = gridDim.x * blockDim.x;
    for (int i = blockIdx.x * blockDim.x + threadIdx.x; i < n; i += stride) p[i] = 0.0f;
}

// ---------------- child-sum scatter: h_tilde[dst] += h[src] ----------------
__global__ void scatter_child_sum_kernel(const int* __restrict__ ei,
                                         const float* __restrict__ h,
                                         float* __restrict__ ht) {
    int e = blockIdx.x;
    int ch = threadIdx.x;
    int src = ei[e];
    int dst = ei[N_EDGES + e];
    atomicAdd(&ht[(size_t)dst * HD + ch], h[(size_t)src * HD + ch]);
}

// ---------------- fused gate GEMM (bf16 WMMA, f32 accum, TDM-staged LDS) ----------------
// Block: 512 thr = 16 waves. One 16-node tile per block; wave w owns channel tile [16w,16w+16).
#define LSTR 264   // 256 + 8 bf16 pad (132 dwords) — matches TDM pad descriptor

__global__ void __launch_bounds__(512)
gate_gemm_kernel(const unsigned short* __restrict__ x_bf,
                 const float* __restrict__ ht,
                 const unsigned short* __restrict__ h_bf,
                 const unsigned short* __restrict__ WB,   // [8][NL][HD][256] bf16
                 const float* __restrict__ Wi_b, const float* __restrict__ Ui_b,
                 const float* __restrict__ Wf_b, const float* __restrict__ Uf_b,
                 const float* __restrict__ Wo_b, const float* __restrict__ Uo_b,
                 const float* __restrict__ Wu_b, const float* __restrict__ Uu_b,
                 int layer,
                 float* __restrict__ c_next, float* __restrict__ o_g,
                 float* __restrict__ wfx, float* __restrict__ ufh) {
    __shared__ unsigned short smem[3 * 16 * LSTR];
    unsigned short* xs  = smem;                 // x tile      (TDM)
    unsigned short* hts = smem + 16 * LSTR;     // h_tilde tile (VALU-converted)
    unsigned short* hs  = smem + 32 * LSTR;     // h tile      (TDM)

    const int tid   = threadIdx.x;
    const int node0 = blockIdx.x * 16;

    // ---- kick off async TDM loads of x and h bf16 tiles (wave 0 issues; EXEC ignored) ----
    if (tid == 0) {
        unsigned int base = (unsigned int)(unsigned long long)(uintptr_t)smem;
        tdm_load_tile_bf16(x_bf + (size_t)node0 * HD, base);
        tdm_load_tile_bf16(h_bf + (size_t)node0 * HD, base + 32u * LSTR * 2u);
    }

    // ---- stage h_tilde (f32 -> bf16) into LDS while TDM runs ----
    {
        int row = tid >> 5;            // 0..15
        int k0  = (tid & 31) * 8;      // 0..248
        const float4* gt = (const float4*)(ht + (size_t)(node0 + row) * HD + k0);
        float4 t0 = gt[0], t1 = gt[1];
        unsigned short* dt = &hts[row * LSTR + k0];
        dt[0]=f2bf(t0.x); dt[1]=f2bf(t0.y); dt[2]=f2bf(t0.z); dt[3]=f2bf(t0.w);
        dt[4]=f2bf(t1.x); dt[5]=f2bf(t1.y); dt[6]=f2bf(t1.z); dt[7]=f2bf(t1.w);
    }
    if (tid == 0) {
        __builtin_amdgcn_s_wait_tensorcnt(0);
    }
    __syncthreads();

    const int wv   = tid >> 5;         // wave id -> channel tile
    const int lane = tid & 31;
    const int n0   = wv * 16;
    const int nn   = lane & 15;
    const int half = lane >> 4;
    const int m    = lane & 15;

    const size_t msz = (size_t)HD * 256;
    const unsigned short* Wi = WB + (size_t)(0 * NL + layer) * msz;
    const unsigned short* Ui = WB + (size_t)(1 * NL + layer) * msz;
    const unsigned short* Wf = WB + (size_t)(2 * NL + layer) * msz;
    const unsigned short* Uf = WB + (size_t)(3 * NL + layer) * msz;
    const unsigned short* Wo = WB + (size_t)(4 * NL + layer) * msz;
    const unsigned short* Uo = WB + (size_t)(5 * NL + layer) * msz;
    const unsigned short* Wu = WB + (size_t)(6 * NL + layer) * msz;
    const unsigned short* Uu = WB + (size_t)(7 * NL + layer) * msz;

    v8f acc_i  = {0,0,0,0,0,0,0,0};
    v8f acc_o  = {0,0,0,0,0,0,0,0};
    v8f acc_u  = {0,0,0,0,0,0,0,0};
    v8f acc_f  = {0,0,0,0,0,0,0,0};
    v8f acc_uf = {0,0,0,0,0,0,0,0};

    #pragma unroll
    for (int kk = 0; kk < 8; ++kk) {
        const int k0 = kk * 32;
        FragBF ax, ah, aho;
        #pragma unroll
        for (int v = 0; v < 8; ++v) {
            int bk = ((v & 4) ? 16 : 0) + half * 8 + (v & 3) * 2;   // ISA 16-bit A layout
            ax.u[v]  = *(const unsigned int*)&xs [m * LSTR + k0 + bk];
            ah.u[v]  = *(const unsigned int*)&hts[m * LSTR + k0 + bk];
            aho.u[v] = *(const unsigned int*)&hs [m * LSTR + k0 + bk];
        }
        FragBF bwi, bui, bwf, buff, bwo, buo, bwu, buu;
        #pragma unroll
        for (int v = 0; v < 8; ++v) {
            int bk = ((v & 4) ? 16 : 0) + half * 8 + (v & 3) * 2;
            size_t off = (size_t)(n0 + nn) * 256 + k0 + bk;
            bwi.u[v]  = *(const unsigned int*)&Wi[off];
            bui.u[v]  = *(const unsigned int*)&Ui[off];
            bwf.u[v]  = *(const unsigned int*)&Wf[off];
            buff.u[v] = *(const unsigned int*)&Uf[off];
            bwo.u[v]  = *(const unsigned int*)&Wo[off];
            buo.u[v]  = *(const unsigned int*)&Uo[off];
            bwu.u[v]  = *(const unsigned int*)&Wu[off];
            buu.u[v]  = *(const unsigned int*)&Uu[off];
        }
        acc_i  = __builtin_amdgcn_wmma_f32_16x16x32_bf16(false, ax.v,  false, bwi.v,  (short)0, acc_i,  false, false);
        acc_i  = __builtin_amdgcn_wmma_f32_16x16x32_bf16(false, ah.v,  false, bui.v,  (short)0, acc_i,  false, false);
        acc_o  = __builtin_amdgcn_wmma_f32_16x16x32_bf16(false, ax.v,  false, bwo.v,  (short)0, acc_o,  false, false);
        acc_o  = __builtin_amdgcn_wmma_f32_16x16x32_bf16(false, ah.v,  false, buo.v,  (short)0, acc_o,  false, false);
        acc_u  = __builtin_amdgcn_wmma_f32_16x16x32_bf16(false, ax.v,  false, bwu.v,  (short)0, acc_u,  false, false);
        acc_u  = __builtin_amdgcn_wmma_f32_16x16x32_bf16(false, ah.v,  false, buu.v,  (short)0, acc_u,  false, false);
        acc_f  = __builtin_amdgcn_wmma_f32_16x16x32_bf16(false, ax.v,  false, bwf.v,  (short)0, acc_f,  false, false);
        acc_uf = __builtin_amdgcn_wmma_f32_16x16x32_bf16(false, aho.v, false, buff.v, (short)0, acc_uf, false, false);
    }

    // ---- epilogue: biases + activations; 16x16 f32 C/D lane layout ----
    const int cidx = layer * HD + n0 + nn;
    const float bi = Wi_b[cidx] + Ui_b[cidx];
    const float bo = Wo_b[cidx] + Uo_b[cidx];
    const float bu = Wu_b[cidx] + Uu_b[cidx];
    const float bf = Wf_b[cidx];
    const float bh = Uf_b[cidx];

    #pragma unroll
    for (int r = 0; r < 8; ++r) {
        int mrow = r + half * 8;
        size_t o = (size_t)(node0 + mrow) * HD + n0 + nn;
        float ig = fmaxf(acc_i[r] + bi, 0.0f);
        float og = fmaxf(acc_o[r] + bo, 0.0f);
        float ug = tanhf(acc_u[r] + bu);
        c_next[o] = ig * ug;
        o_g[o]    = og;
        wfx[o]    = acc_f[r]  + bf;
        ufh[o]    = acc_uf[r] + bh;
    }
}

// ---------------- forget-gate edge scatter ----------------
__global__ void edge_forget_kernel(const int* __restrict__ ei,
                                   const float* __restrict__ wfx,
                                   const float* __restrict__ ufh,
                                   const float* __restrict__ c_old,
                                   float* __restrict__ c_next) {
    int e = blockIdx.x;
    int ch = threadIdx.x;
    int src = ei[e];
    int dst = ei[N_EDGES + e];
    float f = fmaxf(wfx[(size_t)dst * HD + ch] + ufh[(size_t)src * HD + ch], 0.0f);
    atomicAdd(&c_next[(size_t)dst * HD + ch], f * c_old[(size_t)src * HD + ch]);
}

// ---------------- h = o * tanh(c) (+ bf16 mirror of h) ----------------
__global__ void update_h_kernel(const float* __restrict__ o_g,
                                const float* __restrict__ c,
                                float* __restrict__ h,
                                unsigned short* __restrict__ h_bf, int n) {
    int stride = gridDim.x * blockDim.x;
    for (int i = blockIdx.x * blockDim.x + threadIdx.x; i < n; i += stride) {
        float v = o_g[i] * tanhf(c[i]);
        h[i] = v;
        h_bf[i] = f2bf(v);
    }
}

// ---------------- per-graph mean pooling (scatter) ----------------
__global__ void pool_scatter_kernel(const int* __restrict__ batch,
                                    const float* __restrict__ h,
                                    float* __restrict__ sums,
                                    float* __restrict__ cnt) {
    int nIdx = blockIdx.x;
    int ch = threadIdx.x;
    int b = batch[nIdx];
    atomicAdd(&sums[(size_t)b * HD + ch], h[(size_t)nIdx * HD + ch]);
    if (ch == 0) atomicAdd(&cnt[b], 1.0f);
}

// ---------------- stl_feat = hg @ hid_proj^T + b ----------------
__global__ void stl_proj_kernel(const float* __restrict__ sums,
                                const float* __restrict__ cnt,
                                const float* __restrict__ pw,
                                const float* __restrict__ pb,
                                float* __restrict__ out) {
    __shared__ float hg[HD];
    int b = blockIdx.x, t = threadIdx.x;
    float cn = fmaxf(cnt[b], 1.0f);
    hg[t] = sums[(size_t)b * HD + t] / cn;
    __syncthreads();
    float acc = pb[t];
    const float* wr = pw + (size_t)t * HD;
    for (int k = 0; k < HD; ++k) acc += hg[k] * wr[k];
    out[(size_t)b * OUTSTR + t] = acc;
}

// ---------------- ego MLP: 64 -> 256 -> 256 -> 64 ----------------
__global__ void ego_mlp_kernel(const float* __restrict__ ego,
                               const float* __restrict__ w1, const float* __restrict__ b1,
                               const float* __restrict__ w2, const float* __restrict__ b2,
                               const float* __restrict__ w3, const float* __restrict__ b3,
                               float* __restrict__ out) {
    __shared__ float e0[EGOD];
    __shared__ float e1[MHD];
    __shared__ float e2[MHD];
    int b = blockIdx.x, t = threadIdx.x;
    if (t < EGOD) e0[t] = ego[(size_t)b * EGOD + t];
    __syncthreads();
    {
        float acc = b1[t];
        const float* wr = w1 + (size_t)t * EGOD;
        for (int k = 0; k < EGOD; ++k) acc += e0[k] * wr[k];
        e1[t] = fmaxf(acc, 0.0f);
    }
    __syncthreads();
    {
        float acc = b2[t];
        const float* wr = w2 + (size_t)t * MHD;
        for (int k = 0; k < MHD; ++k) acc += e1[k] * wr[k];
        e2[t] = fmaxf(acc, 0.0f);
    }
    __syncthreads();
    if (t < EGOD) {
        float acc = b3[t];
        const float* wr = w3 + (size_t)t * MHD;
        for (int k = 0; k < MHD; ++k) acc += e2[k] * wr[k];
        out[(size_t)b * OUTSTR + CONDD + t] = acc;
    }
}

extern "C" void kernel_launch(void* const* d_in, const int* in_sizes, int n_in,
                              void* d_out, int out_size, void* d_ws, size_t ws_size,
                              hipStream_t stream) {
    (void)in_sizes; (void)n_in; (void)out_size; (void)ws_size;
    const float* x     = (const float*)d_in[0];
    const int*   ei    = (const int*)d_in[1];
    /* d_in[2] = depths (unused: per-level scatters sum to one full scatter) */
    const int*   batch = (const int*)d_in[3];
    const float* ego   = (const float*)d_in[4];
    const float* Wi_w = (const float*)d_in[5];  const float* Wi_b = (const float*)d_in[6];
    const float* Ui_w = (const float*)d_in[7];  const float* Ui_b = (const float*)d_in[8];
    const float* Wf_w = (const float*)d_in[9];  const float* Wf_b = (const float*)d_in[10];
    const float* Uf_w = (const float*)d_in[11]; const float* Uf_b = (const float*)d_in[12];
    const float* Wo_w = (const float*)d_in[13]; const float* Wo_b = (const float*)d_in[14];
    const float* Uo_w = (const float*)d_in[15]; const float* Uo_b = (const float*)d_in[16];
    const float* Wu_w = (const float*)d_in[17]; const float* Wu_b = (const float*)d_in[18];
    const float* Uu_w = (const float*)d_in[19]; const float* Uu_b = (const float*)d_in[20];
    const float* pw = (const float*)d_in[21]; const float* pb = (const float*)d_in[22];
    const float* w1 = (const float*)d_in[23]; const float* b1 = (const float*)d_in[24];
    const float* w2 = (const float*)d_in[25]; const float* b2 = (const float*)d_in[26];
    const float* w3 = (const float*)d_in[27]; const float* b3 = (const float*)d_in[28];

    // ---- workspace carve-up ----
    char* ws = (char*)d_ws;
    const int nW = NL * HD * 256;                       // elements per weight matrix
    unsigned short* WB = (unsigned short*)ws;           // 8 bf16 weight matrices
    ws += (size_t)8 * nW * sizeof(unsigned short);
    unsigned short* x_bf = (unsigned short*)ws; ws += (size_t)NHD * sizeof(unsigned short);
    unsigned short* h_bf = (unsigned short*)ws; ws += (size_t)NHD * sizeof(unsigned short);
    const size_t NHDB = (size_t)NHD * sizeof(float);
    float* h    = (float*)ws; ws += NHDB;
    float* cA   = (float*)ws; ws += NHDB;
    float* cB   = (float*)ws; ws += NHDB;
    float* htld = (float*)ws; ws += NHDB;
    float* og   = (float*)ws; ws += NHDB;
    float* wfxb = (float*)ws; ws += NHDB;
    float* ufhb = (float*)ws; ws += NHDB;
    float* sums = (float*)ws; ws += (size_t)NB * HD * sizeof(float);
    float* cnt  = (float*)ws; ws += (size_t)NB * sizeof(float);

    dim3 t256(256);
    int cvtBlocks = (nW + 255) / 256;
    cvt_bf16_kernel<<<cvtBlocks, t256, 0, stream>>>(Wi_w, WB + (size_t)0 * nW, nW);
    cvt_bf16_kernel<<<cvtBlocks, t256, 0, stream>>>(Ui_w, WB + (size_t)1 * nW, nW);
    cvt_bf16_kernel<<<cvtBlocks, t256, 0, stream>>>(Wf_w, WB + (size_t)2 * nW, nW);
    cvt_bf16_kernel<<<cvtBlocks, t256, 0, stream>>>(Uf_w, WB + (size_t)3 * nW, nW);
    cvt_bf16_kernel<<<cvtBlocks, t256, 0, stream>>>(Wo_w, WB + (size_t)4 * nW, nW);
    cvt_bf16_kernel<<<cvtBlocks, t256, 0, stream>>>(Uo_w, WB + (size_t)5 * nW, nW);
    cvt_bf16_kernel<<<cvtBlocks, t256, 0, stream>>>(Wu_w, WB + (size_t)6 * nW, nW);
    cvt_bf16_kernel<<<cvtBlocks, t256, 0, stream>>>(Uu_w, WB + (size_t)7 * nW, nW);
    cvt_bf16_kernel<<<(NHD + 255) / 256, t256, 0, stream>>>(x, x_bf, NHD);  // x mirror, once

    zero_f32_kernel<<<4096, t256, 0, stream>>>(h, NHD);
    zero_f32_kernel<<<4096, t256, 0, stream>>>(cA, NHD);
    zero_f32_kernel<<<4096, t256, 0, stream>>>((float*)h_bf, NHD / 2);  // bf16 zeros == 0x0000

    float* c_cur = cA;
    float* c_nxt = cB;
    for (int l = 0; l < NL; ++l) {
        zero_f32_kernel<<<4096, t256, 0, stream>>>(htld, NHD);
        scatter_child_sum_kernel<<<N_EDGES, t256, 0, stream>>>(ei, h, htld);
        gate_gemm_kernel<<<N_NODES / 16, dim3(512), 0, stream>>>(
            x_bf, htld, h_bf, WB,
            Wi_b, Ui_b, Wf_b, Uf_b, Wo_b, Uo_b, Wu_b, Uu_b,
            l, c_nxt, og, wfxb, ufhb);
        edge_forget_kernel<<<N_EDGES, t256, 0, stream>>>(ei, wfxb, ufhb, c_cur, c_nxt);
        update_h_kernel<<<8192, t256, 0, stream>>>(og, c_nxt, h, h_bf, NHD);
        float* tmp = c_cur; c_cur = c_nxt; c_nxt = tmp;
    }

    zero_f32_kernel<<<256, t256, 0, stream>>>(sums, NB * HD);
    zero_f32_kernel<<<1, t256, 0, stream>>>(cnt, NB);
    pool_scatter_kernel<<<N_NODES, t256, 0, stream>>>(batch, h, sums, cnt);
    stl_proj_kernel<<<NB, t256, 0, stream>>>(sums, cnt, pw, pb, (float*)d_out);
    ego_mlp_kernel<<<NB, t256, 0, stream>>>(ego, w1, b1, w2, b2, w3, b3, (float*)d_out);
}
